// mamba_block1_10256381903621
// MI455X (gfx1250) — matrix operand order repaired and checked by reading, not compiled, not run on hardware
//
#include <hip/hip_runtime.h>
#include <hip/hip_bf16.h>

typedef __attribute__((ext_vector_type(16))) _Float16 v16h;
typedef __attribute__((ext_vector_type(8)))  float    v8f;

#define D_MODEL 256
#define D_INNER 512
#define NSTATE  64
#define DTRANK  16
#define DTPAD   32
#define DBLCOLS 144
#define DBLPAD  160   // x_proj N padded to a multiple of 32 (10 x 16-tiles)
#define SEQL    1024
#define NBATCH  4
#define NTOK    (NBATCH * SEQL)   // 4096 tokens

// ---------------------------------------------------------------------------
// LayerNorm over d_model=256, one token per 256-thread block, emit f16.
// ---------------------------------------------------------------------------
__global__ void ln_kernel(const float* __restrict__ x, const float* __restrict__ g,
                          const float* __restrict__ bln, _Float16* __restrict__ xn) {
    __shared__ float red[D_MODEL];
    const int row = blockIdx.x;
    const int c   = threadIdx.x;
    const float v = x[(size_t)row * D_MODEL + c];
    red[c] = v; __syncthreads();
    for (int s = D_MODEL / 2; s > 0; s >>= 1) {
        if (c < s) red[c] += red[c + s];
        __syncthreads();
    }
    const float mu = red[0] * (1.f / D_MODEL);
    __syncthreads();
    const float dv = v - mu;
    red[c] = dv * dv; __syncthreads();
    for (int s = D_MODEL / 2; s > 0; s >>= 1) {
        if (c < s) red[c] += red[c + s];
        __syncthreads();
    }
    const float var = red[0] * (1.f / D_MODEL);
    const float xv  = dv * rsqrtf(var + 1e-5f) * g[c] + bln[c];
    xn[(size_t)row * D_MODEL + c] = (_Float16)xv;
}

// ---------------------------------------------------------------------------
// WMMA f16 GEMM: C[M,N] = A[M,K] * W[N,K]^T, f32 accumulate.
// Register-blocked: each wave computes a 64x32 output block (4 M-tiles x 2
// N-tiles) -> 8 v_wmma per K=32 step with 2x A- and 4x B-fragment reuse (~21 FLOP/B).
// REQUIRES M%64==0 and N%32==0 (callers pad N) so the loop body is straight-
// line with EXEC all-ones around every WMMA and no divergent dual-path code
// (the round-2 tail branch caused saveexec predication + VGPR spills).
// mode 0: plain store; mode 1: softplus(val + bias[col]); mode 2: val + res.
// A/B fragments use contiguous K (element s = k0 + 16*(lane/16) + s) in both; C rows 8*(lane/16)+v (wave32).
// ---------------------------------------------------------------------------
#define GEMM_MT 4   // 16-row tiles per wave
#define GEMM_NT 2   // 16-col tiles per wave

__device__ __forceinline__ v16h load_a16(const _Float16* __restrict__ A, int lda,
                                         int row, int k0, int lane) {
    // lane<16: K = k0+[0..15]; lane>=16: K = k0+[16..31]
    const _Float16* p = A + (size_t)row * lda + k0 + ((lane >> 4) << 4);
    v16h f;
#pragma unroll
    for (int j = 0; j < 8; ++j) { f[j] = p[j]; f[j + 8] = p[j + 8]; }
    return f;
}

__device__ __forceinline__ v16h load_b16(const _Float16* __restrict__ W, int ldw,
                                         int col, int k0, int lane) {
    // lane holds column col, 16 contiguous K values starting at k0 + (lane>>4)*16
    const _Float16* p = W + (size_t)col * ldw + k0 + ((lane >> 4) << 4);
    return *(const v16h*)p;
}

__global__ void __launch_bounds__(256)
gemm_f16_wmma(const _Float16* __restrict__ A, const _Float16* __restrict__ W,
              float* __restrict__ C, int M, int N, int K, int mode,
              const float* __restrict__ bias, const float* __restrict__ res) {
    const int lane    = threadIdx.x & 31;
    const int wave    = blockIdx.x * (blockDim.x >> 5) + (threadIdx.x >> 5);
    const int ngroups = N >> 5;                        // 32-col groups (N%32==0)
    const int mgroups = M >> 6;                        // 64-row groups
    if (wave >= mgroups * ngroups) return;             // wave-uniform
    const int mg = wave / ngroups, ng = wave % ngroups;
    const int m0 = mg << 6, n0 = ng << 5;

    const int arow = m0 + (lane & 15);
    const int bcol = n0 + (lane & 15);

    v8f acc[GEMM_MT][GEMM_NT];
#pragma unroll
    for (int i = 0; i < GEMM_MT; ++i)
#pragma unroll
        for (int j = 0; j < GEMM_NT; ++j) acc[i][j] = (v8f){};

    for (int k0 = 0; k0 < K; k0 += 32) {
        v16h a[GEMM_MT];
#pragma unroll
        for (int i = 0; i < GEMM_MT; ++i)
            a[i] = load_a16(A, K, arow + (i << 4), k0, lane);
        v16h b0 = load_b16(W, K, bcol, k0, lane);
        v16h b1 = load_b16(W, K, bcol + 16, k0, lane);
#pragma unroll
        for (int i = 0; i < GEMM_MT; ++i) {
            acc[i][0] = __builtin_amdgcn_wmma_f32_16x16x32_f16(
                false, a[i], false, b0, (short)0, acc[i][0], false, false);
            acc[i][1] = __builtin_amdgcn_wmma_f32_16x16x32_f16(
                false, a[i], false, b1, (short)0, acc[i][1], false, false);
        }
    }

    const int crow0 = m0 + ((lane >> 4) << 3);
    const int ccol0 = n0 + (lane & 15);
#pragma unroll
    for (int i = 0; i < GEMM_MT; ++i) {
#pragma unroll
        for (int j = 0; j < GEMM_NT; ++j) {
            const int ccol = ccol0 + (j << 4);
#pragma unroll
            for (int v = 0; v < 8; ++v) {
                const size_t idx = (size_t)(crow0 + (i << 4) + v) * N + ccol;
                float val = acc[i][j][v];
                if (mode == 1) {                        // dt_proj: bias + softplus
                    val += bias[ccol];
                    val = (val > 20.f) ? val : log1pf(__expf(val));
                } else if (mode == 2) {                 // out_proj: residual add
                    val += res[idx];
                }
                C[idx] = val;
            }
        }
    }
}

// ---------------------------------------------------------------------------
// Causal depthwise conv1d (K=4) + SiLU. xm lives in xz[:, 0:512] (row stride 1024).
// Emits f32 (scan input u) and f16 (x_proj GEMM input).
// ---------------------------------------------------------------------------
__global__ void conv_silu_kernel(const float* __restrict__ xz, const float* __restrict__ cw,
                                 const float* __restrict__ cb, float* __restrict__ xc32,
                                 _Float16* __restrict__ xc16) {
    const int idx = blockIdx.x * blockDim.x + threadIdx.x;
    if (idx >= NTOK * D_INNER) return;
    const int d = idx & (D_INNER - 1);
    const int t = (idx / D_INNER) & (SEQL - 1);
    const int b = idx / (D_INNER * SEQL);
    const float* base = xz + (size_t)(b * SEQL) * (2 * D_INNER);
    float s = cb[d];
#pragma unroll
    for (int k = 0; k < 4; ++k) {
        const int tt = t - 3 + k;
        if (tt >= 0) s += base[(size_t)tt * (2 * D_INNER) + d] * cw[d * 4 + k];
    }
    const float sv = s / (1.f + __expf(-s));     // SiLU
    xc32[idx] = sv;
    xc16[idx] = (_Float16)sv;
}

// ---------------------------------------------------------------------------
// f32 -> f16 convert with zero padding in rows and columns
// (weights, x_proj weight row-pad 144->160, dt input col-pad 16->32).
// ---------------------------------------------------------------------------
__global__ void cvt_pad_f16(const float* __restrict__ src, int srcStride,
                            int srcRows, int srcCols,
                            _Float16* __restrict__ dst, int dstCols, int dstRows) {
    const int idx = blockIdx.x * blockDim.x + threadIdx.x;
    if (idx >= dstRows * dstCols) return;
    const int j = idx % dstCols, r = idx / dstCols;
    const float v = (r < srcRows && j < srcCols) ? src[(size_t)r * srcStride + j] : 0.f;
    dst[idx] = (_Float16)v;
}

// ---------------------------------------------------------------------------
// Selective scan. One workgroup per batch, one thread per channel d.
// State h[64] and -exp(A_log)[64] held in VGPRs; B_t/C_t broadcast via LDS.
// Fused: y = scan + u*Dp; y *= silu(z); emit f16 for out_proj.
// ---------------------------------------------------------------------------
__global__ void __launch_bounds__(D_INNER)
scan_kernel(const float* __restrict__ dt, const float* __restrict__ dbl,
            const float* __restrict__ xc, const float* __restrict__ xz,
            const float* __restrict__ A_log, const float* __restrict__ Dp,
            _Float16* __restrict__ ygate) {
    const int b = blockIdx.x;
    const int d = threadIdx.x;
    __shared__ float sBC[2 * NSTATE];            // B_t[64] then C_t[64]

    float negA[NSTATE], h[NSTATE];
#pragma unroll
    for (int n = 0; n < NSTATE; ++n) {
        negA[n] = -__expf(A_log[(size_t)d * NSTATE + n]);   // A (negative)
        h[n] = 0.f;
    }
    const float Dd = Dp[d];

    for (int t = 0; t < SEQL; ++t) {
        const size_t row = (size_t)b * SEQL + t;
        if (d < 2 * NSTATE) sBC[d] = dbl[row * DBLPAD + DTRANK + d];
        __syncthreads();

        const float dtv = dt[row * D_INNER + d];            // softplus already applied
        const float u   = xc[row * D_INNER + d];
        const float du  = dtv * u;
        float y = 0.f;
#pragma unroll
        for (int n = 0; n < NSTATE; ++n) {
            const float dA = __expf(dtv * negA[n]);
            h[n] = fmaf(dA, h[n], du * sBC[n]);
            y    = fmaf(h[n], sBC[NSTATE + n], y);
        }
        const float zv = xz[row * (2 * D_INNER) + D_INNER + d];
        const float g  = zv / (1.f + __expf(-zv));          // silu(z)
        ygate[row * D_INNER + d] = (_Float16)((y + u * Dd) * g);
        __syncthreads();                                    // protect sBC for next t
    }
}

// ---------------------------------------------------------------------------
// Host-side orchestration
// ---------------------------------------------------------------------------
static inline int ceil_div(int a, int b) { return (a + b - 1) / b; }

static inline int gemm_blocks(int M, int N) {
    const int ngroups = N >> 5;          // N must be a multiple of 32
    const int mgroups = M >> 6;
    return ceil_div(mgroups * ngroups, 8);   // 8 waves per 256-thread block
}

extern "C" void kernel_launch(void* const* d_in, const int* in_sizes, int n_in,
                              void* d_out, int out_size, void* d_ws, size_t ws_size,
                              hipStream_t stream) {
    (void)in_sizes; (void)n_in; (void)out_size; (void)ws_size;
    const float* x_in   = (const float*)d_in[0];
    const float* ln_g   = (const float*)d_in[1];
    const float* ln_b   = (const float*)d_in[2];
    const float* in_w   = (const float*)d_in[3];
    const float* conv_w = (const float*)d_in[4];
    const float* conv_b = (const float*)d_in[5];
    const float* xp_w   = (const float*)d_in[6];
    const float* dt_w   = (const float*)d_in[7];
    const float* dt_b   = (const float*)d_in[8];
    const float* A_log  = (const float*)d_in[9];
    const float* Dp     = (const float*)d_in[10];
    const float* out_w  = (const float*)d_in[11];

    char* ws = (char*)d_ws;
    size_t off = 0;
    auto alloc = [&](size_t bytes) -> void* {
        void* p = ws + off;
        off = (off + bytes + 255) & ~(size_t)255;
        return p;
    };
    _Float16* xn16   = (_Float16*)alloc((size_t)NTOK * D_MODEL * 2);
    float*    xz32   = (float*)   alloc((size_t)NTOK * 2 * D_INNER * 4);
    float*    xc32   = (float*)   alloc((size_t)NTOK * D_INNER * 4);
    _Float16* xc16   = (_Float16*)alloc((size_t)NTOK * D_INNER * 2);
    float*    dbl32  = (float*)   alloc((size_t)NTOK * DBLPAD * 4);   // padded N=160
    _Float16* dtin16 = (_Float16*)alloc((size_t)NTOK * DTPAD * 2);
    float*    dt32   = (float*)   alloc((size_t)NTOK * D_INNER * 4);
    _Float16* yg16   = (_Float16*)alloc((size_t)NTOK * D_INNER * 2);
    float*    xbuf   = (float*)   alloc((size_t)NTOK * D_MODEL * 4);
    _Float16* w_in   = (_Float16*)alloc((size_t)2 * D_INNER * D_MODEL * 2);
    _Float16* w_xp   = (_Float16*)alloc((size_t)DBLPAD * D_INNER * 2);  // row-padded
    _Float16* w_dt   = (_Float16*)alloc((size_t)D_INNER * DTPAD * 2);
    _Float16* w_out  = (_Float16*)alloc((size_t)D_MODEL * D_INNER * 2);

    const int depth = 2;
    for (int l = 0; l < depth; ++l) {
        const float* g_l   = ln_g   + (size_t)l * D_MODEL;
        const float* b_l   = ln_b   + (size_t)l * D_MODEL;
        const float* inw_l = in_w   + (size_t)l * 2 * D_INNER * D_MODEL;
        const float* cw_l  = conv_w + (size_t)l * D_INNER * 4;
        const float* cb_l  = conv_b + (size_t)l * D_INNER;
        const float* xpw_l = xp_w   + (size_t)l * DBLCOLS * D_INNER;
        const float* dtw_l = dt_w   + (size_t)l * D_INNER * DTRANK;
        const float* dtb_l = dt_b   + (size_t)l * D_INNER;
        const float* al_l  = A_log  + (size_t)l * D_INNER * NSTATE;
        const float* dp_l  = Dp     + (size_t)l * D_INNER;
        const float* ow_l  = out_w  + (size_t)l * D_MODEL * D_INNER;

        const float* xcur  = (l == 0) ? x_in : xbuf;
        float*       xnext = (l == 0) ? xbuf : (float*)d_out;

        // Weight conversions to f16:
        //   xp_w row-padded 144 -> 160, dt_w col-padded K 16 -> 32.
        cvt_pad_f16<<<ceil_div(2 * D_INNER * D_MODEL, 256), 256, 0, stream>>>(
            inw_l, D_MODEL, 2 * D_INNER, D_MODEL, w_in, D_MODEL, 2 * D_INNER);
        cvt_pad_f16<<<ceil_div(DBLPAD * D_INNER, 256), 256, 0, stream>>>(
            xpw_l, D_INNER, DBLCOLS, D_INNER, w_xp, D_INNER, DBLPAD);
        cvt_pad_f16<<<ceil_div(D_INNER * DTPAD, 256), 256, 0, stream>>>(
            dtw_l, DTRANK, D_INNER, DTRANK, w_dt, DTPAD, D_INNER);
        cvt_pad_f16<<<ceil_div(D_MODEL * D_INNER, 256), 256, 0, stream>>>(
            ow_l, D_INNER, D_MODEL, D_INNER, w_out, D_INNER, D_MODEL);

        // 1) LayerNorm -> xn16
        ln_kernel<<<NTOK, D_MODEL, 0, stream>>>(xcur, g_l, b_l, xn16);

        // 2) in_proj: [4096,256] x [1024,256]^T -> xz32 [4096,1024]
        gemm_f16_wmma<<<gemm_blocks(NTOK, 2 * D_INNER), 256, 0, stream>>>(
            xn16, w_in, xz32, NTOK, 2 * D_INNER, D_MODEL, 0, nullptr, nullptr);

        // 3) causal depthwise conv + SiLU
        conv_silu_kernel<<<ceil_div(NTOK * D_INNER, 256), 256, 0, stream>>>(
            xz32, cw_l, cb_l, xc32, xc16);

        // 4) x_proj: [4096,512] x [160,512]^T -> dbl32 [4096,160] (cols 144..159 junk)
        gemm_f16_wmma<<<gemm_blocks(NTOK, DBLPAD), 256, 0, stream>>>(
            xc16, w_xp, dbl32, NTOK, DBLPAD, D_INNER, 0, nullptr, nullptr);

        // 5) pack dt slice (first 16 cols of dbl) into padded f16 [4096,32]
        cvt_pad_f16<<<ceil_div(NTOK * DTPAD, 256), 256, 0, stream>>>(
            dbl32, DBLPAD, NTOK, DTRANK, dtin16, DTPAD, NTOK);

        // 6) dt_proj + bias + softplus: [4096,32] x [512,32]^T -> dt32 [4096,512]
        gemm_f16_wmma<<<gemm_blocks(NTOK, D_INNER), 256, 0, stream>>>(
            dtin16, w_dt, dt32, NTOK, D_INNER, DTPAD, 1, dtb_l, nullptr);

        // 7) selective scan + D-skip + SiLU gating -> yg16
        scan_kernel<<<NBATCH, D_INNER, 0, stream>>>(
            dt32, dbl32, xc32, xz32, al_l, dp_l, yg16);

        // 8) out_proj + residual: [4096,512] x [256,512]^T + xcur -> xnext
        gemm_f16_wmma<<<gemm_blocks(NTOK, D_MODEL), 256, 0, stream>>>(
            yg16, w_out, xnext, NTOK, D_MODEL, D_INNER, 2, nullptr, xcur);
    }
}
